// densityPropGRUCell_second_56092272885819
// MI455X (gfx1250) — matrix operand order repaired and checked by reading, not compiled, not run on hardware
//
#include <hip/hip_runtime.h>
#include <hip/hip_bf16.h>

// ---------------------------------------------------------------------------
// densityPropGRUCell for MI455X (gfx1250).
// Dominant cost: 6 congruences A^T S A = 12 batched 1024^3 GEMMs (412 GFLOP)
// -> v_wmma_f32_16x16x32_f16, f32 accumulate.  Memory footprint ~350 MB
// (~15us @ 23.3 TB/s) so the WMMA pipe is the bottleneck: big wave tiles
// (64x64 = 16 WMMAs/wave/slab), coalesced stores only (no transposed global
// writes), LDS double buffering.
// ---------------------------------------------------------------------------

typedef __attribute__((ext_vector_type(16))) _Float16 v16h;
typedef __attribute__((ext_vector_type(8)))  float    v8f;
typedef __attribute__((ext_vector_type(2)))  _Float16 h2;

constexpr int BATCH = 16;
constexpr int DIN   = 1024;
constexpr int DU    = 1024;

__device__ __forceinline__ float sigmoidf_(float x) { return 1.0f / (1.0f + __expf(-x)); }
__device__ __forceinline__ float softplusf_(float x) { return (x > 20.0f) ? x : log1pf(__expf(x)); }

// ---------------------------------------------------------------------------
// WMMA GEMM:  C[m,n] (+)= sum_k Aop[k,m]*B[k,n]
//   aKMajor==1 : A stored A[k*lda + m]   (TN form, A = weight)
//   aKMajor==0 : A stored A[m*lda + k]   (NN form, A = pass-1 result)
//   B always   : B[k*ldb + n]
//   accum  ==1 : accumulators preloaded from C (beta = 1)
// Block tile 128x256x32, 256 threads = 8 waves (2x4), wave tile 64x64
// = 4x4 v_wmma_f32_16x16x32_f16.  Double-buffered LDS (2 x ~30KB).
// ---------------------------------------------------------------------------
#define BM 128
#define BN 256
#define BKK 32
#define LDA_S (BKK + 8)
#define LDB_S (BKK + 8)

__device__ __forceinline__
void stage_tiles(const float* __restrict__ A, int lda, int aKMajor,
                 const float* __restrict__ B, int ldb,
                 int k0, int tileM, int tileN, int tid,
                 _Float16* __restrict__ As, _Float16* __restrict__ Bs)
{
    if (aKMajor) {
        // A[k][m], m contiguous: 128 cols x 32 rows = 1024 float4
#pragma unroll
        for (int i = 0; i < 4; ++i) {
            const int f    = tid * 4 + i;
            const int krow = f >> 5;           // 0..31
            const int c4   = (f & 31) * 4;     // 0..124
            const float4 av = *(const float4*)(A + (long)(k0 + krow) * lda + tileM + c4);
            As[(c4 + 0) * LDA_S + krow] = (_Float16)av.x;
            As[(c4 + 1) * LDA_S + krow] = (_Float16)av.y;
            As[(c4 + 2) * LDA_S + krow] = (_Float16)av.z;
            As[(c4 + 3) * LDA_S + krow] = (_Float16)av.w;
        }
    } else {
        // A[m][k], k contiguous: 128 rows x 8 float4
#pragma unroll
        for (int i = 0; i < 4; ++i) {
            const int f  = tid * 4 + i;
            const int m  = f >> 3;             // 0..127
            const int k4 = (f & 7) * 4;        // 0..28
            const float4 av = *(const float4*)(A + (long)(tileM + m) * lda + k0 + k4);
            As[m * LDA_S + k4 + 0] = (_Float16)av.x;
            As[m * LDA_S + k4 + 1] = (_Float16)av.y;
            As[m * LDA_S + k4 + 2] = (_Float16)av.z;
            As[m * LDA_S + k4 + 3] = (_Float16)av.w;
        }
    }
    // B[k][n], n contiguous: 32 rows x 64 float4 = 2048 float4
#pragma unroll
    for (int i = 0; i < 8; ++i) {
        const int f    = tid * 8 + i;
        const int krow = f >> 6;               // 0..31
        const int c4   = (f & 63) * 4;         // 0..252
        const float4 bv = *(const float4*)(B + (long)(k0 + krow) * ldb + tileN + c4);
        Bs[(c4 + 0) * LDB_S + krow] = (_Float16)bv.x;
        Bs[(c4 + 1) * LDB_S + krow] = (_Float16)bv.y;
        Bs[(c4 + 2) * LDB_S + krow] = (_Float16)bv.z;
        Bs[(c4 + 3) * LDB_S + krow] = (_Float16)bv.w;
    }
}

__global__ __launch_bounds__(256)
void wmma_gemm_kernel(const float* __restrict__ A, long strideA, int lda, int aKMajor,
                      const float* __restrict__ B, long strideB, int ldb,
                      float* __restrict__ C, long strideC, int ldc,
                      int K, int accum)
{
    __shared__ _Float16 As[2][BM * LDA_S];
    __shared__ _Float16 Bs[2][BN * LDB_S];

    A += (long)blockIdx.z * strideA;
    B += (long)blockIdx.z * strideB;
    C += (long)blockIdx.z * strideC;

    const int tileM = blockIdx.x * BM;
    const int tileN = blockIdx.y * BN;

    const int tid    = threadIdx.x;
    const int lane   = tid & 31;
    const int wave   = tid >> 5;
    const int waveM  = wave & 1;    // 2 waves along M (64 rows each)
    const int waveN  = wave >> 1;   // 4 waves along N (64 cols each)
    const int lanelo = lane & 15;
    const int khalf  = lane >> 4;

    v8f acc[4][4];
#pragma unroll
    for (int mt = 0; mt < 4; ++mt)
#pragma unroll
        for (int nt = 0; nt < 4; ++nt) {
            if (accum) {
#pragma unroll
                for (int rr = 0; rr < 8; ++rr) {
                    const int m = tileM + waveM * 64 + mt * 16 + rr + khalf * 8;
                    const int n = tileN + waveN * 64 + nt * 16 + lanelo;
                    acc[mt][nt][rr] = C[(long)m * ldc + n];
                }
            } else {
                v8f zero = {0.f, 0.f, 0.f, 0.f, 0.f, 0.f, 0.f, 0.f};
                acc[mt][nt] = zero;
            }
        }

    const int nIter = K / BKK;
    stage_tiles(A, lda, aKMajor, B, ldb, 0, tileM, tileN, tid, As[0], Bs[0]);

    for (int it = 0; it < nIter; ++it) {
        const int p = it & 1;
        __syncthreads();   // staging of buf p complete; compute on buf 1-p done

        if (it + 1 < nIter)
            stage_tiles(A, lda, aKMajor, B, ldb, (it + 1) * BKK,
                        tileM, tileN, tid, As[p ^ 1], Bs[p ^ 1]);
        if (it + 2 < nIter) {  // L2 prefetch two slabs ahead (global_prefetch_b8)
            const int kp = (it + 2) * BKK;
            __builtin_prefetch(B + (long)(kp + (tid >> 3)) * ldb + tileN + (tid & 7) * 32, 0, 0);
        }

        // ---- fragments per documented wave32 16-bit WMMA layouts ----------
        const _Float16* __restrict__ Ap = As[p];
        const _Float16* __restrict__ Bp = Bs[p];
        v16h afrag[4];
#pragma unroll
        for (int mt = 0; mt < 4; ++mt) {
            const int m = waveM * 64 + mt * 16 + lanelo;
#pragma unroll
            for (int j = 0; j < 8; ++j) {
                // A 16x32: lanes 0-15 K(0..7,16..23), lanes 16-31 K(8..15,24..31)
                const int bk = ((j & 4) << 2) + khalf * 8 + (j & 3) * 2;
                h2 pr = *(const h2*)&Ap[m * LDA_S + bk];
                afrag[mt][2 * j]     = pr[0];
                afrag[mt][2 * j + 1] = pr[1];
            }
        }
        v16h bfrag[4];
#pragma unroll
        for (int nt = 0; nt < 4; ++nt) {
            const int n    = waveN * 64 + nt * 16 + lanelo;
            const int koff = khalf * 8;   // B 32x16: lanes 0-15 K(0..7,16..23), lanes 16-31 K(8..15,24..31)
#pragma unroll
            for (int j = 0; j < 8; ++j) {
                h2 pr = *(const h2*)&Bp[n * LDB_S + ((j & 4) << 2) + koff + (j & 3) * 2];
                bfrag[nt][2 * j]     = pr[0];
                bfrag[nt][2 * j + 1] = pr[1];
            }
        }

        // ---- 16 WMMAs per wave per K-slab ---------------------------------
#pragma unroll
        for (int mt = 0; mt < 4; ++mt)
#pragma unroll
            for (int nt = 0; nt < 4; ++nt)
                acc[mt][nt] = __builtin_amdgcn_wmma_f32_16x16x32_f16(
                    false, afrag[mt], false, bfrag[nt],
                    (short)0, acc[mt][nt], false, false);
    }

    // ---- coalesced writeback (D: vgpr rr -> M=rr(+8 hi lanes), N=lane&15) --
#pragma unroll
    for (int mt = 0; mt < 4; ++mt)
#pragma unroll
        for (int nt = 0; nt < 4; ++nt)
#pragma unroll
            for (int rr = 0; rr < 8; ++rr) {
                const int m = tileM + waveM * 64 + mt * 16 + rr + khalf * 8;
                const int n = tileN + waveN * 64 + nt * 16 + lanelo;
                C[(long)m * ldc + n] = acc[mt][nt][rr];
            }
}

// ------------------------- reductions (per batch) --------------------------
__global__ __launch_bounds__(256)
void sumsq_rows_kernel(const float* __restrict__ x, float* __restrict__ out, int N)
{
    __shared__ float red[256];
    const int b = blockIdx.x;
    float s = 0.f;
    for (int i = threadIdx.x; i < N; i += 256) {
        const float v = x[(long)b * N + i];
        s += v * v;
    }
    red[threadIdx.x] = s; __syncthreads();
    for (int off = 128; off > 0; off >>= 1) {
        if (threadIdx.x < off) red[threadIdx.x] += red[threadIdx.x + off];
        __syncthreads();
    }
    if (threadIdx.x == 0) out[b] = red[0];
}

__global__ __launch_bounds__(256)
void trace_kernel(const float* __restrict__ S, float* __restrict__ out, int N)
{
    __shared__ float red[256];
    const int b = blockIdx.x;
    float s = 0.f;
    for (int i = threadIdx.x; i < N; i += 256)
        s += S[(long)b * N * N + (long)i * N + i];
    red[threadIdx.x] = s; __syncthreads();
    for (int off = 128; off > 0; off >>= 1) {
        if (threadIdx.x < off) red[threadIdx.x] += red[threadIdx.x + off];
        __syncthreads();
    }
    if (threadIdx.x == 0) out[b] = red[0];
}

// --------------------------- gate pre-activations --------------------------
__global__ __launch_bounds__(256)
void gates_zr_kernel(const float* __restrict__ mu, const float* __restrict__ prev,
                     const float* __restrict__ Uz, const float* __restrict__ Wz,
                     const float* __restrict__ Ur, const float* __restrict__ Wr,
                     float* __restrict__ z, float* __restrict__ r)
{
    const int u = blockIdx.x * 256 + threadIdx.x;
    const int b = blockIdx.y;
    float az = 0.f, ar = 0.f;
    for (int i = 0; i < DIN; ++i) {
        const float m = mu[(long)b * DIN + i];
        az += m * Uz[(long)i * DU + u];
        ar += m * Ur[(long)i * DU + u];
    }
    for (int j = 0; j < DU; ++j) {
        const float p = prev[(long)b * DU + j];
        az += p * Wz[(long)j * DU + u];
        ar += p * Wr[(long)j * DU + u];
    }
    z[(long)b * DU + u] = sigmoidf_(az);
    r[(long)b * DU + u] = sigmoidf_(ar);
}

__global__ __launch_bounds__(256)
void gate_h_kernel(const float* __restrict__ mu, const float* __restrict__ sr,
                   const float* __restrict__ Uh, const float* __restrict__ Wh,
                   float* __restrict__ h)
{
    const int u = blockIdx.x * 256 + threadIdx.x;
    const int b = blockIdx.y;
    float a = 0.f;
    for (int i = 0; i < DIN; ++i) a += mu[(long)b * DIN + i] * Uh[(long)i * DU + u];
    for (int j = 0; j < DU;  ++j) a += sr[(long)b * DU + j] * Wh[(long)j * DU + u];
    h[(long)b * DU + u] = tanhf(a);
}

__global__ __launch_bounds__(256)
void sr_kernel(const float* __restrict__ prev, const float* __restrict__ r,
               float* __restrict__ sr, int n)
{
    const int i = blockIdx.x * 256 + threadIdx.x;
    if (i < n) sr[i] = prev[i] * r[i];
}

// --------- Sigma_gate = g'(u) g'(v) * (acc + diagonal xSx/trS terms) -------
__global__ __launch_bounds__(256)
void gate_sigma_kernel(const float* __restrict__ acc, const float* __restrict__ g,
                       const float* __restrict__ usig, const float* __restrict__ wsig,
                       const float* __restrict__ xx1, const float* __restrict__ tr1,
                       const float* __restrict__ xx2, const float* __restrict__ tr2,
                       float* __restrict__ out, int mode /*0: z(1-z), 1: 1-h^2*/)
{
    const int idx = blockIdx.x * 256 + threadIdx.x;   // < DU*DU
    const int b   = blockIdx.y;
    const int u = idx >> 10, v = idx & (DU - 1);
    const float gu = g[(long)b * DU + u];
    const float gv = g[(long)b * DU + v];
    const float du = (mode == 0) ? gu * (1.f - gu) : (1.f - gu * gu);
    const float dv = (mode == 0) ? gv * (1.f - gv) : (1.f - gv * gv);
    float a = acc[(long)b * DU * DU + idx];
    if (u == v)
        a += (xx1[b] + tr1[b]) * softplusf_(usig[u]) +
             (xx2[b] + tr2[b]) * softplusf_(wsig[u]);
    out[(long)b * DU * DU + idx] = du * dv * a;
}

// ---- sigma_g = Ss*Sr + prev_u prev_v * Sr + r_u r_v * Ss  (in place on Sr) --
__global__ __launch_bounds__(256)
void sigma_g_kernel(const float* __restrict__ Ss, float* __restrict__ Sr,
                    const float* __restrict__ prev, const float* __restrict__ r)
{
    const int idx = blockIdx.x * 256 + threadIdx.x;
    const int b   = blockIdx.y;
    const int u = idx >> 10, v = idx & (DU - 1);
    const long o = (long)b * DU * DU + idx;
    const float ss = Ss[o], sr_ = Sr[o];
    Sr[o] = ss * sr_
          + prev[(long)b * DU + u] * prev[(long)b * DU + v] * sr_
          + r[(long)b * DU + u]   * r[(long)b * DU + v]   * ss;
}

// ------------------------------- final combine -----------------------------
__global__ __launch_bounds__(256)
void mu_out_kernel(const float* __restrict__ z, const float* __restrict__ prev,
                   const float* __restrict__ h, float* __restrict__ mu_out, int n)
{
    const int i = blockIdx.x * 256 + threadIdx.x;
    if (i < n) mu_out[i] = z[i] * prev[i] + (1.f - z[i]) * h[i];
}

__global__ __launch_bounds__(256)
void combine_kernel(const float* __restrict__ Sz, const float* __restrict__ Sh,
                    const float* __restrict__ Ss, const float* __restrict__ z,
                    const float* __restrict__ prev, const float* __restrict__ h,
                    float* __restrict__ Sout)
{
    const int idx = blockIdx.x * 256 + threadIdx.x;
    const int b   = blockIdx.y;
    const int u = idx >> 10, v = idx & (DU - 1);
    const long o = (long)b * DU * DU + idx;
    const float sz = Sz[o], sh = Sh[o], ss = Ss[o];
    const float zu = z[(long)b * DU + u],    zv = z[(long)b * DU + v];
    const float pu = prev[(long)b * DU + u], pv = prev[(long)b * DU + v];
    const float hu = h[(long)b * DU + u],    hv = h[(long)b * DU + v];
    float s = sz * ss + zu * zv * ss + pu * pv * sz                  // sigma_a
            + sz * sh + (1.f - zu) * (1.f - zv) * sh + hu * hv * sz  // sigma_b
            - sz * pu * hv                                           // sigma_ab
            - sz * hu * pv;                                          // sigma_abT
    if (!isfinite(s)) s = 0.f;
    if (u == v) s = fabsf(s);
    Sout[o] = s;
}

// ---------------------------------------------------------------------------
extern "C" void kernel_launch(void* const* d_in, const int* in_sizes, int n_in,
                              void* d_out, int out_size, void* d_ws, size_t ws_size,
                              hipStream_t stream)
{
    (void)in_sizes; (void)n_in; (void)out_size; (void)ws_size;
    const float* mu_in  = (const float*)d_in[0];
    const float* sig_in = (const float*)d_in[1];
    const float* prev   = (const float*)d_in[2];
    const float* Ss     = (const float*)d_in[3];
    const float* U_z = (const float*)d_in[4];  const float* uz_s = (const float*)d_in[5];
    const float* W_z = (const float*)d_in[6];  const float* wz_s = (const float*)d_in[7];
    const float* U_r = (const float*)d_in[8];  const float* ur_s = (const float*)d_in[9];
    const float* W_r = (const float*)d_in[10]; const float* wr_s = (const float*)d_in[11];
    const float* U_h = (const float*)d_in[12]; const float* uh_s = (const float*)d_in[13];
    const float* W_h = (const float*)d_in[14]; const float* wh_s = (const float*)d_in[15];

    float* mu_out  = (float*)d_out;
    float* Sig_out = (float*)d_out + (long)BATCH * DU;

    const long MSZ = (long)BATCH * DU * DU;
    float* w   = (float*)d_ws;
    float* tmp = w; w += MSZ;           // pass-1 result P = A^T S (row-major)
    float* acc = w; w += MSZ;           // congruence accumulator (aliased as Sigma_h)
    float* Sz  = w; w += MSZ;
    float* Sr  = w; w += MSZ;           // Sigma_r, becomes sigma_g in place
    float* zv  = w; w += (long)BATCH * DU;
    float* rv  = w; w += (long)BATCH * DU;
    float* hv  = w; w += (long)BATCH * DU;
    float* srv = w; w += (long)BATCH * DU;
    float* xx_mu   = w; w += BATCH;
    float* xx_prev = w; w += BATCH;
    float* xx_sr   = w; w += BATCH;
    float* tr_in   = w; w += BATCH;
    float* tr_st   = w; w += BATCH;
    float* tr_g    = w; w += BATCH;
    float* Sh = acc;  // Sigma_h written in place over acc

    const dim3 ggrid(DU / BM, DU / BN, BATCH);
    const dim3 eg(DU * DU / 256, BATCH);
    auto gemm = [&](const float* Aop, long sA, int aKMajor, const float* Bop, long sB,
                    float* Cop, int accum) {
        wmma_gemm_kernel<<<ggrid, 256, 0, stream>>>(
            Aop, sA, DU, aKMajor, Bop, sB, DU, Cop, (long)DU * DU, DU, DU, accum);
    };

    // batch scalars
    sumsq_rows_kernel<<<BATCH, 256, 0, stream>>>(mu_in, xx_mu, DIN);
    sumsq_rows_kernel<<<BATCH, 256, 0, stream>>>(prev,  xx_prev, DU);
    trace_kernel<<<BATCH, 256, 0, stream>>>(sig_in, tr_in, DIN);
    trace_kernel<<<BATCH, 256, 0, stream>>>(Ss,     tr_st, DU);

    // gate means
    gates_zr_kernel<<<dim3(DU / 256, BATCH), 256, 0, stream>>>(mu_in, prev, U_z, W_z, U_r, W_r, zv, rv);
    sr_kernel<<<(BATCH * DU) / 256, 256, 0, stream>>>(prev, rv, srv, BATCH * DU);
    sumsq_rows_kernel<<<BATCH, 256, 0, stream>>>(srv, xx_sr, DU);
    gate_h_kernel<<<dim3(DU / 256, BATCH), 256, 0, stream>>>(mu_in, srv, U_h, W_h, hv);

    // ---- z gate: acc = U_z^T sig_in U_z + W_z^T Ss W_z -> Sigma_z ----------
    gemm(U_z, 0, 1, sig_in, (long)DIN * DIN, tmp, 0);   // tmp = U_z^T S   (TN)
    gemm(tmp, (long)DU * DU, 0, U_z, 0, acc, 0);        // acc = tmp U_z   (NN)
    gemm(W_z, 0, 1, Ss, (long)DU * DU, tmp, 0);
    gemm(tmp, (long)DU * DU, 0, W_z, 0, acc, 1);
    gate_sigma_kernel<<<eg, 256, 0, stream>>>(acc, zv, uz_s, wz_s, xx_mu, tr_in, xx_prev, tr_st, Sz, 0);

    // ---- r gate -> Sigma_r -------------------------------------------------
    gemm(U_r, 0, 1, sig_in, (long)DIN * DIN, tmp, 0);
    gemm(tmp, (long)DU * DU, 0, U_r, 0, acc, 0);
    gemm(W_r, 0, 1, Ss, (long)DU * DU, tmp, 0);
    gemm(tmp, (long)DU * DU, 0, W_r, 0, acc, 1);
    gate_sigma_kernel<<<eg, 256, 0, stream>>>(acc, rv, ur_s, wr_s, xx_mu, tr_in, xx_prev, tr_st, Sr, 0);

    // ---- sigma_g (in place over Sigma_r) + its trace -----------------------
    sigma_g_kernel<<<eg, 256, 0, stream>>>(Ss, Sr, prev, rv);
    trace_kernel<<<BATCH, 256, 0, stream>>>(Sr, tr_g, DU);

    // ---- h gate: acc = U_h^T sig_in U_h + W_h^T sigma_g W_h -> Sigma_h -----
    gemm(U_h, 0, 1, sig_in, (long)DIN * DIN, tmp, 0);
    gemm(tmp, (long)DU * DU, 0, U_h, 0, acc, 0);
    gemm(W_h, 0, 1, Sr, (long)DU * DU, tmp, 0);
    gemm(tmp, (long)DU * DU, 0, W_h, 0, acc, 1);
    gate_sigma_kernel<<<eg, 256, 0, stream>>>(acc, hv, uh_s, wh_s, xx_mu, tr_in, xx_sr, tr_g, Sh, 1);

    // ---- outputs -----------------------------------------------------------
    mu_out_kernel<<<(BATCH * DU) / 256, 256, 0, stream>>>(zv, prev, hv, mu_out, BATCH * DU);
    combine_kernel<<<eg, 256, 0, stream>>>(Sz, Sh, Ss, zv, prev, hv, Sig_out);
}
